// SplitModel_35588099014903
// MI455X (gfx1250) — compile-verified, matrix operand order checked by reading, not run on hardware
//
#include <hip/hip_runtime.h>
#include <hip/hip_bf16.h>
#include <math.h>
#include <stdint.h>

// ---------------------------------------------------------------------------
// Model dims (compile-time constants from the reference)
// ---------------------------------------------------------------------------
#define Bz   16
#define Lz   512
#define Hz   256
#define Vz   32000
#define HALF 128
#define TWO_H 512

typedef __bf16 bf16_t;
typedef __attribute__((ext_vector_type(16))) __bf16 v16bf;
typedef __attribute__((ext_vector_type(8)))  float  v8f;

// ---------------------------------------------------------------------------
// CDNA5 async global->LDS helpers (ASYNCcnt path, ISA §10 / §15.18.3).
// LDS byte address = low 32 bits of the generic pointer (flat->LDS aperture
// truncation, ISA §10.2). GV addressing form: vDst(lds), v[addr64], off.
// ---------------------------------------------------------------------------
__device__ __forceinline__ uint32_t lds_addr_of(const void* p) {
    return (uint32_t)(size_t)p;
}
__device__ __forceinline__ void async_load_b128(uint32_t lds_dst, const void* gsrc) {
    asm volatile("global_load_async_to_lds_b128 %0, %1, off"
                 :: "v"(lds_dst), "v"((unsigned long long)(size_t)gsrc)
                 : "memory");
}
__device__ __forceinline__ void wait_async0() {
    asm volatile("s_wait_asynccnt 0x0" ::: "memory");
}

// ---------------------------------------------------------------------------
// WMMA fragment loaders (bf16 16x16x32, wave32; layouts per ISA 7.12.2)
// lane = hi*16 + r holds row r; elements e<8 -> K=k0+hi*8+e, e>=8 -> +16.
// Same pattern serves A (rows of X) and B (rows of W, since D = X @ W^T).
// ---------------------------------------------------------------------------
__device__ __forceinline__ v16bf load_frag_cvt(const float* src, int ld, int k0) {
    const int lane = threadIdx.x & 31;
    const float* row = src + (size_t)(lane & 15) * ld + k0 + (lane >> 4) * 8;
    v16bf f;
#pragma unroll
    for (int e = 0; e < 8; ++e) f[e]     = (bf16_t)row[e];
#pragma unroll
    for (int e = 0; e < 8; ++e) f[8 + e] = (bf16_t)row[16 + e];
    return f;
}
__device__ __forceinline__ v16bf load_frag_raw(const bf16_t* src, int ld, int k0) {
    const int lane = threadIdx.x & 31;
    const bf16_t* row = src + (size_t)(lane & 15) * ld + k0 + (lane >> 4) * 8;
    v16bf f;
#pragma unroll
    for (int e = 0; e < 8; ++e) f[e]     = row[e];
#pragma unroll
    for (int e = 0; e < 8; ++e) f[8 + e] = row[16 + e];
    return f;
}
__device__ __forceinline__ v8f wmma_bf16(v16bf a, v16bf b, v8f c) {
    return __builtin_amdgcn_wmma_f32_16x16x32_bf16(
        false, a, false, b, (short)0, c, false, false);
}

// ---------------------------------------------------------------------------
// Kernel W: one-time fp32 -> bf16 weight conversion (amortized over 512 blocks
// of kernel A; W_out is intentionally NOT converted since it is single-use).
// ---------------------------------------------------------------------------
__global__ __launch_bounds__(256)
void cvt_f32_bf16(const float* __restrict__ src, bf16_t* __restrict__ dst, int n) {
    const int i = blockIdx.x * 256 + threadIdx.x;
    if (i < n) dst[i] = (bf16_t)src[i];
}

// ---------------------------------------------------------------------------
// Kernel A: embed gather (async->LDS) + MLP + residual + LayerNorm + k-projs
// One block = 16 flattened tokens; 256 threads = 8 waves; 40 KB LDS.
// ---------------------------------------------------------------------------
__global__ __launch_bounds__(256)
void tok_mlp_ln_proj(const int* __restrict__ seq, const float* __restrict__ embed,
                     const bf16_t* __restrict__ W1b, const float* __restrict__ b1,
                     const bf16_t* __restrict__ W2b, const float* __restrict__ b2,
                     const float* __restrict__ gamma, const float* __restrict__ beta,
                     const bf16_t* __restrict__ Wsemb, const bf16_t* __restrict__ Wepib,
                     float* __restrict__ ksg, float* __restrict__ keg)
{
    __shared__ float  sX [16 * Hz];     // fp32: h -> x -> (LN scratch)   16 KB
    __shared__ bf16_t sXb[16 * Hz];     // bf16 A operand (h, then LN'd)   8 KB
    __shared__ bf16_t sHb[16 * TWO_H];  // bf16 relu hidden               16 KB

    const int tid  = threadIdx.x;
    const int wave = tid >> 5;
    const int lane = tid & 31;
    const int hi   = lane >> 4;
    const int c    = lane & 15;
    const int t0   = blockIdx.x * 16;

    // ---- async gather: 16 embedding rows straight into LDS (b128 chunks) ----
    for (int i = tid; i < (16 * Hz) / 4; i += 256) {
        const int tok = i >> 6;          // 64 float4 per 256-wide row
        const int c4  = i & 63;
        async_load_b128(lds_addr_of(&sX[tok * Hz + c4 * 4]),
                        embed + (size_t)seq[t0 + tok] * Hz + c4 * 4);
    }
    wait_async0();
    __syncthreads();
    for (int i = tid; i < 16 * Hz; i += 256) sXb[i] = (bf16_t)sX[i];
    __syncthreads();

    // ---- GEMM1: hidden = relu(h @ W1^T + b1); N=512 -> 32 tiles, 4/wave ----
    for (int nt = 0; nt < 4; ++nt) {
        const int n0 = (wave * 4 + nt) * 16;
        v8f acc = {};
#pragma unroll
        for (int kt = 0; kt < Hz / 32; ++kt) {
            acc = wmma_bf16(load_frag_raw(sXb, Hz, kt * 32),
                            load_frag_raw(W1b + (size_t)n0 * Hz, Hz, kt * 32), acc);
        }
        const int n = n0 + c;
        const float bias = b1[n];
#pragma unroll
        for (int r8 = 0; r8 < 8; ++r8)
            sHb[(r8 + hi * 8) * TWO_H + n] = (bf16_t)fmaxf(acc[r8] + bias, 0.0f);
    }
    __syncthreads();

    // ---- GEMM2 + bias + residual: x = h + hidden @ W2^T + b2; 2 tiles/wave ----
    for (int nt = 0; nt < 2; ++nt) {
        const int n0 = (wave * 2 + nt) * 16;
        v8f acc = {};
#pragma unroll
        for (int kt = 0; kt < TWO_H / 32; ++kt) {
            acc = wmma_bf16(load_frag_raw(sHb, TWO_H, kt * 32),
                            load_frag_raw(W2b + (size_t)n0 * TWO_H, TWO_H, kt * 32), acc);
        }
        const int n = n0 + c;
        const float bias = b2[n];
#pragma unroll
        for (int r8 = 0; r8 < 8; ++r8) {
            const int m = r8 + hi * 8;
            sX[m * Hz + n] = sX[m * Hz + n] + acc[r8] + bias;
        }
    }
    __syncthreads();

    // ---- LayerNorm per token row (2 rows/wave); bf16 result -> sXb ----
    for (int rr = 0; rr < 2; ++rr) {
        const int row = wave * 2 + rr;
        float s = 0.f, ss = 0.f;
        for (int j = lane; j < Hz; j += 32) {
            const float x = sX[row * Hz + j];
            s += x; ss += x * x;
        }
#pragma unroll
        for (int m = 16; m; m >>= 1) {
            s  += __shfl_xor(s,  m, 32);
            ss += __shfl_xor(ss, m, 32);
        }
        const float mu   = s * (1.0f / Hz);
        const float var  = ss * (1.0f / Hz) - mu * mu;
        const float rstd = rsqrtf(var + 1e-5f);
        for (int j = lane; j < Hz; j += 32) {
            const float x = (sX[row * Hz + j] - mu) * rstd;
            sXb[row * Hz + j] = (bf16_t)(x * gamma[j] + beta[j]);
        }
    }
    __syncthreads();

    // ---- Projections: ks = hn @ Wsem^T, ke = hn @ Wepi^T; 2 tiles/wave ----
    for (int p = 0; p < 2; ++p) {
        const int ntile = wave * 2 + p;                  // 0..15
        const bf16_t* W = (ntile < 8) ? Wsemb : Wepib;
        float* outp     = (ntile < 8) ? ksg : keg;       // [B*L, HALF] fp32
        const int n0    = (ntile & 7) * 16;
        v8f acc = {};
#pragma unroll
        for (int kt = 0; kt < Hz / 32; ++kt) {
            acc = wmma_bf16(load_frag_raw(sXb, Hz, kt * 32),
                            load_frag_raw(W + (size_t)n0 * Hz, Hz, kt * 32), acc);
        }
        const int n = n0 + c;
#pragma unroll
        for (int r8 = 0; r8 < 8; ++r8)
            outp[(size_t)(t0 + r8 + hi * 8) * HALF + n] = acc[r8];
    }
}

// ---------------------------------------------------------------------------
// Kernel B: delta-rule scan. 32 blocks = (batch, {sem,epi}); 256 threads.
// State M[128][128] fp32 in registers (64/thread). Per-step key is async-
// loaded into LDS by wave 0 (ASYNCcnt), normalized, then rank-1 updated.
// Inherently serial over L-1 steps -> VALU, not WMMA.
// ---------------------------------------------------------------------------
__global__ __launch_bounds__(256)
void delta_scan(const float* __restrict__ ksg, const float* __restrict__ keg,
                float* __restrict__ cbuf /* [B, 2*HALF] */)
{
    const int b   = blockIdx.x >> 1;
    const int mem = blockIdx.x & 1;
    const float* kall = mem ? keg : ksg;      // [B*L, HALF]

    const int t     = threadIdx.x;
    const int row   = t >> 1;                 // 0..127
    const int cbase = (t & 1) * 64;

    float M[64];
#pragma unroll
    for (int j = 0; j < 64; ++j) M[j] = 0.0f;

    __shared__ float sk[HALF];
    __shared__ float s_rinv;

    for (int l = 0; l < Lz - 1; ++l) {
        const float* kv = kall + ((size_t)b * Lz + l) * HALF;
        if (t < 32) {                          // wave 0 stages the key via async DMA
            async_load_b128(lds_addr_of(&sk[t * 4]), kv + t * 4);
            wait_async0();
        }
        __syncthreads();
        if (t < 32) {                          // wave 0: 1/max(||k||, eps)
            float s = 0.f;
#pragma unroll
            for (int j = 0; j < 4; ++j) { const float v = sk[t * 4 + j]; s += v * v; }
#pragma unroll
            for (int m = 16; m; m >>= 1) s += __shfl_xor(s, m, 32);
            if (t == 0) s_rinv = 1.0f / fmaxf(sqrtf(s), 1e-12f);
        }
        __syncthreads();
        const float rinv = s_rinv;

        float acc = 0.f;
#pragma unroll
        for (int j = 0; j < 64; ++j) acc += M[j] * sk[cbase + j];
        acc *= rinv;                                        // half row of M@kn
        const float v = acc + __shfl_xor(acc, 1, 32);       // full dot (lane pair)
        const float delta = sk[row] - v;                    // raw k minus read
#pragma unroll
        for (int j = 0; j < 64; ++j) M[j] += delta * (sk[cbase + j] * rinv);
        __syncthreads();                                    // before sk overwrite
    }

    // final read with raw (unnormalized) last-token key
    const float* kv = kall + ((size_t)b * Lz + (Lz - 1)) * HALF;
    if (t < 32) {
        async_load_b128(lds_addr_of(&sk[t * 4]), kv + t * 4);
        wait_async0();
    }
    __syncthreads();
    float acc = 0.f;
#pragma unroll
    for (int j = 0; j < 64; ++j) acc += M[j] * sk[cbase + j];
    const float v = acc + __shfl_xor(acc, 1, 32);
    if ((t & 1) == 0)
        cbuf[b * (2 * HALF) + mem * HALF + row] = v;
}

// ---------------------------------------------------------------------------
// Kernel C: r = c @ W_rp^T + b_rp ([16x256]@[256x256]); writes bf16 r for the
// vocab GEMM. One block, 8 waves, 2 N-tiles each.
// ---------------------------------------------------------------------------
__global__ __launch_bounds__(256)
void r_proj(const float* __restrict__ cbuf, const bf16_t* __restrict__ Wrpb,
            const float* __restrict__ brp, bf16_t* __restrict__ rbufb)
{
    const int wave = threadIdx.x >> 5;
    const int lane = threadIdx.x & 31;
    const int hi = lane >> 4, c = lane & 15;
    for (int nt = 0; nt < 2; ++nt) {
        const int n0 = (wave * 2 + nt) * 16;
        v8f acc = {};
#pragma unroll
        for (int kt = 0; kt < Hz / 32; ++kt) {
            acc = wmma_bf16(load_frag_cvt(cbuf, Hz, kt * 32),
                            load_frag_raw(Wrpb + (size_t)n0 * Hz, Hz, kt * 32), acc);
        }
        const int n = n0 + c;
        const float bias = brp[n];
#pragma unroll
        for (int r8 = 0; r8 < 8; ++r8)
            rbufb[(r8 + hi * 8) * Hz + n] = (bf16_t)(acc[r8] + bias);
    }
}

// ---------------------------------------------------------------------------
// Kernel D: out = r @ W_out^T + b_out ([16x256]@[256x32000]).
// 2000 N-tiles; 250 blocks x 8 waves. W_out is single-use -> fp32 read with
// in-register bf16 conversion (pre-converting would add HBM traffic).
// ---------------------------------------------------------------------------
__global__ __launch_bounds__(256)
void out_gemm(const bf16_t* __restrict__ rbufb, const float* __restrict__ Wout,
              const float* __restrict__ bout, float* __restrict__ out)
{
    const int wave = threadIdx.x >> 5;
    const int lane = threadIdx.x & 31;
    const int hi = lane >> 4, c = lane & 15;
    const int n0 = (blockIdx.x * 8 + wave) * 16;

    v8f acc = {};
#pragma unroll
    for (int kt = 0; kt < Hz / 32; ++kt) {
        acc = wmma_bf16(load_frag_raw(rbufb, Hz, kt * 32),
                        load_frag_cvt(Wout + (size_t)n0 * Hz, Hz, kt * 32), acc);
    }
    const int n = n0 + c;
    const float bias = bout[n];
#pragma unroll
    for (int r8 = 0; r8 < 8; ++r8)
        out[(size_t)(r8 + hi * 8) * Vz + n] = acc[r8] + bias;
}

// ---------------------------------------------------------------------------
// Launch
// ---------------------------------------------------------------------------
extern "C" void kernel_launch(void* const* d_in, const int* in_sizes, int n_in,
                              void* d_out, int out_size, void* d_ws, size_t ws_size,
                              hipStream_t stream) {
    const int*   seq   = (const int*)  d_in[0];
    const float* embed = (const float*)d_in[1];
    const float* W1    = (const float*)d_in[2];
    const float* b1    = (const float*)d_in[3];
    const float* W2    = (const float*)d_in[4];
    const float* b2    = (const float*)d_in[5];
    const float* gamma = (const float*)d_in[6];
    const float* beta  = (const float*)d_in[7];
    const float* Wsem  = (const float*)d_in[8];
    const float* Wepi  = (const float*)d_in[9];
    const float* Wrp   = (const float*)d_in[10];
    const float* brp   = (const float*)d_in[11];
    const float* Wout  = (const float*)d_in[12];
    const float* bout  = (const float*)d_in[13];
    float* out = (float*)d_out;

    // workspace carve-up
    char* w = (char*)d_ws;
    size_t off = 0;
    float*  ksg   = (float*) (w + off); off += (size_t)Bz * Lz * HALF * 4;  // 4 MB
    float*  keg   = (float*) (w + off); off += (size_t)Bz * Lz * HALF * 4;  // 4 MB
    float*  cbuf  = (float*) (w + off); off += (size_t)Bz * 2 * HALF * 4;
    bf16_t* rbufb = (bf16_t*)(w + off); off += (size_t)Bz * Hz * 2;
    bf16_t* W1b   = (bf16_t*)(w + off); off += (size_t)TWO_H * Hz * 2;
    bf16_t* W2b   = (bf16_t*)(w + off); off += (size_t)Hz * TWO_H * 2;
    bf16_t* Wsemb = (bf16_t*)(w + off); off += (size_t)HALF * Hz * 2;
    bf16_t* Wepib = (bf16_t*)(w + off); off += (size_t)HALF * Hz * 2;
    bf16_t* Wrpb  = (bf16_t*)(w + off); off += (size_t)Hz * Hz * 2;

    // one-time weight conversion (multi-use weights only)
    cvt_f32_bf16<<<(TWO_H * Hz + 255) / 256, 256, 0, stream>>>(W1, W1b, TWO_H * Hz);
    cvt_f32_bf16<<<(Hz * TWO_H + 255) / 256, 256, 0, stream>>>(W2, W2b, Hz * TWO_H);
    cvt_f32_bf16<<<(HALF * Hz + 255) / 256, 256, 0, stream>>>(Wsem, Wsemb, HALF * Hz);
    cvt_f32_bf16<<<(HALF * Hz + 255) / 256, 256, 0, stream>>>(Wepi, Wepib, HALF * Hz);
    cvt_f32_bf16<<<(Hz * Hz + 255) / 256, 256, 0, stream>>>(Wrp, Wrpb, Hz * Hz);

    tok_mlp_ln_proj<<<(Bz * Lz) / 16, 256, 0, stream>>>(
        seq, embed, W1b, b1, W2b, b2, gamma, beta, Wsemb, Wepib, ksg, keg);
    delta_scan<<<Bz * 2, 256, 0, stream>>>(ksg, keg, cbuf);
    r_proj<<<1, 256, 0, stream>>>(cbuf, Wrpb, brp, rbufb);
    out_gemm<<<Vz / (16 * 8), 256, 0, stream>>>(rbufb, Wout, bout, out);
}
